// RNNLayer_32195074851537
// MI455X (gfx1250) — matrix operand order repaired and checked by reading, not compiled, not run on hardware
//
#include <hip/hip_runtime.h>
#include <hip/hip_bf16.h>
#include <math.h>

// ---------------------------------------------------------------------------
// RNN layer (tanh), S=1024, B=64, I=H=1024, fp32 in/out, bf16 WMMA internally.
// Phase 1: x_proj GEMM (65536x1024x1024) -> written directly into d_out,
//          B-tile staged into LDS by the Tensor Data Mover (double-buffered).
// Phase 2: persistent 16-WG scan kernel, atomic grid barrier per timestep.
// ---------------------------------------------------------------------------

typedef __attribute__((ext_vector_type(16))) __bf16       v16bf;
typedef __attribute__((ext_vector_type(8)))  float        v8f;
typedef __attribute__((ext_vector_type(4)))  unsigned int v4u;
typedef __attribute__((ext_vector_type(8)))  unsigned int v8u;
typedef __attribute__((ext_vector_type(8)))  int          v8i;
typedef __attribute__((ext_vector_type(4)))  int          v4i;

union FragU { v8u u; v16bf b; };

static __device__ __forceinline__ v16bf make_frag(v4u x0, v4u x1) {
    FragU f;
    f.u[0] = x0[0]; f.u[1] = x0[1]; f.u[2] = x0[2]; f.u[3] = x0[3];
    f.u[4] = x1[0]; f.u[5] = x1[1]; f.u[6] = x1[2]; f.u[7] = x1[3];
    return f.b;
}

// round-to-nearest-even fp32 -> bf16 (packed pair / single)
static __device__ __forceinline__ unsigned packbf2(float a, float b) {
    unsigned ua = __float_as_uint(a);
    unsigned ub = __float_as_uint(b);
    ua += 0x7FFFu + ((ua >> 16) & 1u);
    ub += 0x7FFFu + ((ub >> 16) & 1u);
    return (ua >> 16) | (ub & 0xFFFF0000u);
}
static __device__ __forceinline__ unsigned short packbf1(float a) {
    unsigned ua = __float_as_uint(a);
    ua += 0x7FFFu + ((ua >> 16) & 1u);
    return (unsigned short)(ua >> 16);
}

// native CDNA5 tanh if the toolchain exposes it
#if __has_builtin(__builtin_amdgcn_tanhf)
#define TANH_F32(x) __builtin_amdgcn_tanhf(x)
#elif __has_builtin(__builtin_amdgcn_tanh_f32)
#define TANH_F32(x) __builtin_amdgcn_tanh_f32(x)
#else
#define TANH_F32(x) tanhf(x)
#endif

// Problem dims
constexpr int kS = 1024;   // sequence length
constexpr int kB = 64;     // batch
constexpr int kH = 1024;   // hidden == input dim
constexpr int kMtot = kS * kB;  // 65536 rows in phase-1 GEMM

// Workspace layout (bytes)
constexpr size_t WS_BIAS  = 0;                                  // 1024 f32
constexpr size_t WS_WIH   = 4 * 1024;                           // 1024x1024 bf16
constexpr size_t WS_WHH   = WS_WIH + 2u * 1024 * 1024;          // 1024x1024 bf16
constexpr size_t WS_HPING = WS_WHH + 2u * 1024 * 1024;          // 64x1024 bf16
constexpr size_t WS_HPONG = WS_HPING + 64u * 1024 * 2;          // 64x1024 bf16
constexpr size_t WS_CTR   = WS_HPONG + 64u * 1024 * 2;          // barrier counter

// ---------------------------------------------------------------------------
// Tensor Data Mover: async 2D bf16 tile load, global -> LDS.
// Tile rows are 128B (64 bf16); TDM pads 16B after each row so the LDS image
// has a 72-short row stride (36 dwords -> bank-conflict-free ds_load_b128).
// ---------------------------------------------------------------------------
static __device__ __forceinline__ void tdm_load_tile_bf16(
    unsigned lds_byte_off, const unsigned short* gsrc,
    unsigned tile_d0, unsigned tile_d1,
    unsigned tensor_d0, unsigned tensor_d1, unsigned stride0) {
    unsigned long long ga = (unsigned long long)(const void*)gsrc;

    v4u g0;
    g0[0] = 1u;                                   // count=1 (valid descriptor)
    g0[1] = lds_byte_off;                         // LDS destination offset
    g0[2] = (unsigned)ga;                         // global_addr[31:0]
    g0[3] = ((unsigned)(ga >> 32) & 0x01FFFFFFu)  // global_addr[56:32]
            | (2u << 30);                         // type = 2 ("image")

    v8i g1;
    g1[0] = (int)((1u << 16)     // data_size = 2 bytes
                | (1u << 20)     // pad_enable
                | (4u << 22)     // pad_interval: 32 dwords (one 128B row)
                | (3u << 25));   // pad_amount:   4 dwords (16B)
    g1[1] = (int)((tensor_d0 & 0xFFFFu) << 16);                    // td0[15:0]
    g1[2] = (int)((tensor_d0 >> 16) | ((tensor_d1 & 0xFFFFu) << 16));
    g1[3] = (int)((tensor_d1 >> 16) | (tile_d0 << 16));            // tile_dim0
    g1[4] = (int)(tile_d1 & 0xFFFFu);                              // tile_dim1
    g1[5] = (int)stride0;                                          // stride0 lo
    g1[6] = 0;                                                     // stride0 hi
    g1[7] = 0;

    v4i g2; g2[0] = 1; g2[1] = 1; g2[2] = 0; g2[3] = 0;  // dims 2/3 degenerate
    v4i g3; g3[0] = 0; g3[1] = (int)(1u << 16); g3[2] = 0; g3[3] = 0;

#if __clang_major__ >= 23
    v8i g4 = {0, 0, 0, 0, 0, 0, 0, 0};
    __builtin_amdgcn_tensor_load_to_lds(g0, g1, g2, g3, g4, 0);
#else
    __builtin_amdgcn_tensor_load_to_lds(g0, g1, g2, g3, 0);
#endif
}

// ---------------------------------------------------------------------------
// Kernel 0: fold biases, convert weights + hx to bf16, reset barrier counter.
// ---------------------------------------------------------------------------
__global__ void rnn_prep_kernel(const float* __restrict__ wih,
                                const float* __restrict__ whh,
                                const float* __restrict__ bih,
                                const float* __restrict__ bhh,
                                const float* __restrict__ hx,
                                unsigned char* __restrict__ ws) {
    float*    bias  = (float*)(ws + WS_BIAS);
    unsigned* wihb  = (unsigned*)(ws + WS_WIH);
    unsigned* whhb  = (unsigned*)(ws + WS_WHH);
    unsigned* hping = (unsigned*)(ws + WS_HPING);
    unsigned* ctr   = (unsigned*)(ws + WS_CTR);

    size_t tid = (size_t)blockIdx.x * blockDim.x + threadIdx.x;
    size_t nth = (size_t)gridDim.x * blockDim.x;

    if (tid == 0) *ctr = 0u;

    for (size_t i = tid; i < (size_t)kH; i += nth)
        bias[i] = bih[i] + bhh[i];

    for (size_t i = tid; i < (size_t)kH * kH / 2; i += nth)
        wihb[i] = packbf2(wih[2 * i], wih[2 * i + 1]);

    for (size_t i = tid; i < (size_t)kH * kH / 2; i += nth)
        whhb[i] = packbf2(whh[2 * i], whh[2 * i + 1]);

    for (size_t i = tid; i < (size_t)kB * kH / 2; i += nth)
        hping[i] = packbf2(hx[2 * i], hx[2 * i + 1]);
}

// ---------------------------------------------------------------------------
// Kernel 1: x_proj GEMM.  C[m,n] = sum_k input[m,k]*Wih[n,k] + bias[n]
// Tile: 128x128, BK=64, 256 threads (8 waves), each wave 32x64 (2x4 WMMA tiles)
// A tile: fp32->bf16 VALU conversion into LDS.
// B tile: TDM async DMA into LDS, double-buffered across K iterations.
// ---------------------------------------------------------------------------
constexpr int BM  = 128;
constexpr int BN  = 128;
constexpr int BK  = 64;
constexpr int LDA = BK + 8;  // 72 shorts/row == TDM row (32 dwords) + pad (4)

__global__ void __launch_bounds__(256)
rnn_xproj_kernel(const float* __restrict__ input,
                 const unsigned char* __restrict__ ws,
                 float* __restrict__ out) {
    const unsigned short* wihb = (const unsigned short*)(ws + WS_WIH);
    const float*          bias = (const float*)(ws + WS_BIAS);

    __shared__ unsigned short smem[(BM + 2 * BN) * LDA];
    unsigned short* As  = smem;                 // 128 x 72
    unsigned short* Bs0 = smem + BM * LDA;      // 128 x 72 (TDM buffer 0)
    unsigned short* Bs1 = Bs0 + BN * LDA;       // 128 x 72 (TDM buffer 1)

    // LDS byte offsets: flat LDS-aperture pointers carry the offset in [31:0]
    const unsigned ldsB0 = (unsigned)(size_t)(const void*)Bs0;
    const unsigned ldsB1 = (unsigned)(size_t)(const void*)Bs1;

    const unsigned tid  = threadIdx.x;
    const unsigned lane = tid & 31u;
    const unsigned wid  = tid >> 5;
    const unsigned wm   = wid & 3u;   // 0..3 : 32-row strip
    const unsigned wn   = wid >> 2;   // 0..1 : 64-col strip
    const unsigned l16  = lane & 15u;
    const unsigned half = lane >> 4;

    const unsigned m0 = blockIdx.x * BM;
    const unsigned n0 = blockIdx.y * BN;

    v8f zero = {0.f, 0.f, 0.f, 0.f, 0.f, 0.f, 0.f, 0.f};
    v8f acc[2][4];
#pragma unroll
    for (int i = 0; i < 2; ++i)
#pragma unroll
        for (int j = 0; j < 4; ++j) acc[i][j] = zero;

    const unsigned row = tid >> 1;          // 0..127
    const unsigned kh  = (tid & 1u) * 32u;  // 0 or 32

    // preload B tile 0 via TDM (EXEC is ignored; issue once, from wave 0)
    if (wid == 0)
        tdm_load_tile_bf16(ldsB0, wihb + (size_t)n0 * kH,
                           BK, BN, kH, kH, kH);

    const int kIters = kH / BK;  // 16
    for (int it = 0; it < kIters; ++it) {
        const unsigned kt = (unsigned)it * BK;
        __syncthreads();  // everyone done reading As and B[(it+1)&1]

        // kick off next B tile into the other buffer while we compute this one
        if (wid == 0 && it + 1 < kIters)
            tdm_load_tile_bf16((it & 1) ? ldsB0 : ldsB1,
                               wihb + (size_t)n0 * kH + (kt + BK),
                               BK, BN, kH, kH, kH);

        // --- stage A (fp32 -> bf16) ---
        const float* ga = input + (size_t)(m0 + row) * kH + kt + kh;
#pragma unroll
        for (int j = 0; j < 8; ++j) {
            float4 v = ((const float4*)ga)[j];
            unsigned long long p =
                (unsigned long long)packbf2(v.x, v.y) |
                ((unsigned long long)packbf2(v.z, v.w) << 32);
            *(unsigned long long*)&As[row * LDA + kh + 4 * j] = p;
        }

        // current B tile's DMA must be complete (next tile may stay in flight)
        if (wid == 0) {
            if (it + 1 < kIters) __builtin_amdgcn_s_wait_tensorcnt(1);
            else                 __builtin_amdgcn_s_wait_tensorcnt(0);
        }
        __syncthreads();  // publish A stores + TDM-written B tile

        const unsigned short* Bs = (it & 1) ? Bs1 : Bs0;
#pragma unroll
        for (int ks = 0; ks < BK; ks += 32) {
            // A fragments: lane l16 holds row m; half selects K 0-7/8-15 (+16)
            v16bf af[2];
#pragma unroll
            for (int i = 0; i < 2; ++i) {
                const unsigned short* p =
                    &As[(wm * 32 + i * 16 + l16) * LDA + ks + half * 8];
                af[i] = make_frag(*(const v4u*)p, *(const v4u*)(p + 16));
            }
            // B fragments: lane l16 holds col n; half selects K 0-7/8-15 (+16)
            v16bf bf[4];
#pragma unroll
            for (int j = 0; j < 4; ++j) {
                const unsigned short* p =
                    &Bs[(wn * 64 + j * 16 + l16) * LDA + ks + half * 8];
                bf[j] = make_frag(*(const v4u*)p, *(const v4u*)(p + 16));
            }
#pragma unroll
            for (int i = 0; i < 2; ++i)
#pragma unroll
                for (int j = 0; j < 4; ++j)
                    acc[i][j] = __builtin_amdgcn_wmma_f32_16x16x32_bf16(
                        false, af[i], false, bf[j], (short)0, acc[i][j],
                        false, false);
        }
    }

    // epilogue: + bias, store fp32
#pragma unroll
    for (int i = 0; i < 2; ++i) {
#pragma unroll
        for (int j = 0; j < 4; ++j) {
            unsigned col  = n0 + wn * 64 + j * 16 + l16;
            float    bv   = bias[col];
            unsigned mrow = m0 + wm * 32 + i * 16 + half * 8;
            float* po = out + (size_t)mrow * kH + col;
#pragma unroll
            for (int r = 0; r < 8; ++r)
                po[(size_t)r * kH] = acc[i][j][r] + bv;
        }
    }
}

// ---------------------------------------------------------------------------
// Kernel 2: persistent recurrent scan.  16 WGs, WG w owns columns [64w,64w+64).
// Per step: 64x64 += h(64x1024) * Whh_slice^T, tanh(+x_proj), grid barrier.
// ---------------------------------------------------------------------------
__global__ void __launch_bounds__(256)
rnn_scan_kernel(float* __restrict__ out,      // (S,B,H) fp32, x_proj on entry
                float* __restrict__ hn,       // (B,H) fp32
                unsigned char* __restrict__ ws) {
    const unsigned short* whhb = (const unsigned short*)(ws + WS_WHH);
    unsigned short* hbuf0 = (unsigned short*)(ws + WS_HPING);
    unsigned short* hbuf1 = (unsigned short*)(ws + WS_HPONG);
    unsigned* ctr = (unsigned*)(ws + WS_CTR);

    const unsigned tid  = threadIdx.x;
    const unsigned lane = tid & 31u;
    const unsigned wid  = tid >> 5;
    const unsigned wm   = wid & 3u;   // 16-row strip of batch (0..3)
    const unsigned wn   = wid >> 2;   // 32-col strip (0..1)
    const unsigned l16  = lane & 15u;
    const unsigned half = lane >> 4;
    const unsigned n0   = blockIdx.x * 64u;
    const unsigned mrowbase = wm * 16u;

    v8f zero = {0.f, 0.f, 0.f, 0.f, 0.f, 0.f, 0.f, 0.f};

    for (unsigned s = 0; s < (unsigned)kS; ++s) {
        const unsigned short* hc  = (s & 1u) ? hbuf1 : hbuf0;
        unsigned short*       hnx = (s & 1u) ? hbuf0 : hbuf1;

        // hint x_proj rows toward the caches (global_prefetch_b8)
        __builtin_prefetch(out + ((size_t)s * kB + mrowbase) * kH + n0, 0, 1);

        v8f acc[2] = {zero, zero};
        for (unsigned kt = 0; kt < (unsigned)kH; kt += 32) {
            const unsigned short* pa =
                hc + (size_t)(mrowbase + l16) * kH + kt + half * 8;
            v16bf af = make_frag(*(const v4u*)pa, *(const v4u*)(pa + 16));
#pragma unroll
            for (int j = 0; j < 2; ++j) {
                const unsigned short* pb =
                    whhb + (size_t)(n0 + wn * 32 + j * 16 + l16) * kH + kt +
                    half * 8;
                v16bf bfv = make_frag(*(const v4u*)pb, *(const v4u*)(pb + 16));
                acc[j] = __builtin_amdgcn_wmma_f32_16x16x32_bf16(
                    false, af, false, bfv, (short)0, acc[j], false, false);
            }
        }

        // epilogue: tanh(x_proj + recurrent), write fp32 out + bf16 next-h
#pragma unroll
        for (int j = 0; j < 2; ++j) {
            unsigned col = n0 + wn * 32 + j * 16 + l16;
            unsigned m   = mrowbase + half * 8;
            float*          px = out + ((size_t)s * kB + m) * kH + col;
            unsigned short* ph = hnx + (size_t)m * kH + col;
#pragma unroll
            for (int r = 0; r < 8; ++r) {
                float v = TANH_F32(px[(size_t)r * kH] + acc[j][r]);
                px[(size_t)r * kH] = v;
                ph[(size_t)r * kH] = packbf1(v);
                if (s == (unsigned)kS - 1)
                    hn[(size_t)(m + r) * kH + col] = v;
            }
        }

        // grid barrier (16 resident WGs): release stores, arrive, acquire spin
        __threadfence();
        __syncthreads();
        if (tid == 0) {
            __hip_atomic_fetch_add(ctr, 1u, __ATOMIC_ACQ_REL,
                                   __HIP_MEMORY_SCOPE_AGENT);
            const unsigned target = 16u * (s + 1u);
            while (__hip_atomic_load(ctr, __ATOMIC_ACQUIRE,
                                     __HIP_MEMORY_SCOPE_AGENT) < target) {
                __builtin_amdgcn_s_sleep(2);
            }
        }
        __syncthreads();
        __threadfence();
    }
}

// ---------------------------------------------------------------------------
extern "C" void kernel_launch(void* const* d_in, const int* in_sizes, int n_in,
                              void* d_out, int out_size, void* d_ws,
                              size_t ws_size, hipStream_t stream) {
    (void)in_sizes; (void)n_in; (void)out_size; (void)ws_size;

    const float* input = (const float*)d_in[0];  // (S,B,I)
    const float* hx    = (const float*)d_in[1];  // (B,H)
    const float* wih   = (const float*)d_in[2];  // (H,I)
    const float* whh   = (const float*)d_in[3];  // (H,H)
    const float* bih   = (const float*)d_in[4];  // (H,)
    const float* bhh   = (const float*)d_in[5];  // (H,)

    float* out = (float*)d_out;                       // (S,B,H)
    float* hn  = out + (size_t)kS * kB * kH;          // (B,H)
    unsigned char* ws = (unsigned char*)d_ws;

    rnn_prep_kernel<<<1024, 256, 0, stream>>>(wih, whh, bih, bhh, hx, ws);

    dim3 g1(kMtot / BM, kH / BN);
    rnn_xproj_kernel<<<g1, 256, 0, stream>>>(input, ws, out);

    rnn_scan_kernel<<<16, 256, 0, stream>>>(out, hn, ws);
}
